// AttentionMechanism_68547678044469
// MI455X (gfx1250) — matrix operand order repaired and checked by reading, not compiled, not run on hardware
//
#include <hip/hip_runtime.h>

typedef __attribute__((ext_vector_type(16))) _Float16 v16h;
typedef __attribute__((ext_vector_type(8)))  _Float16 v8h;
typedef __attribute__((ext_vector_type(8)))  float    v8f;
typedef unsigned int u32x4 __attribute__((ext_vector_type(4)));
typedef int          i32x4 __attribute__((ext_vector_type(4)));
typedef int          i32x8 __attribute__((ext_vector_type(8)));

#define B_   4
#define T_   4096
#define D_   128
// scores already in log2 domain: Q pre-scaled by (1/sqrt(D)) * log2(e)
#define QSCALE (0.08838834764831845f * 1.4426950408889634f)

// LDS strides (halfs), padded by TDM pad feature for bank-conflict-free reads
#define KSTRIDE 136   /* 128 + 8  (row = 64 DW, pad 4 DW)  */
#define VSTRIDE 40    /* 32  + 8  (row = 16 DW, pad 4 DW)  */
#define PSTRIDE 40

// ---- A-operand loader: 16-bit A 16x32 per-lane K order is interleaved:
//      lanes 0-15:  K = {0..7, 16..23};  lanes 16-31: K = {8..15, 24..31}
__device__ __forceinline__ v16h load_a16(const _Float16* chunk, int h) {
  v8h lo = *(const v8h*)(chunk + h * 8);
  v8h hi = *(const v8h*)(chunk + 16 + h * 8);
  v16h r;
#pragma unroll
  for (int i = 0; i < 8; i++) { r[i] = lo[i]; r[8 + i] = hi[i]; }
  return r;
}

// ---- B-operand loader: 16-bit B 32x16 per-lane K order is interleaved (matches A):
//      lanes 0-15:  K = {0..7, 16..23};  lanes 16-31: K = {8..15, 24..31}
__device__ __forceinline__ v16h load_b16(const _Float16* chunk, int h) {
  v8h lo = *(const v8h*)(chunk + h * 8);
  v8h hi = *(const v8h*)(chunk + 16 + h * 8);
  v16h r;
#pragma unroll
  for (int i = 0; i < 8; i++) { r[i] = lo[i]; r[8 + i] = hi[i]; }
  return r;
}

// ---- TDM: 2-D tile load, tensor dims == tile dims (never OOB), with LDS row padding.
//      dims/strides in elements (data_size = 2 bytes). D# packed per ISA 8.3/8.4.
__device__ __forceinline__ void tdm_load_2d(unsigned lds_off, const void* gaddr,
                                            unsigned d0, unsigned d1, unsigned stride0,
                                            unsigned pad_interval_code, unsigned pad_amount_code) {
  unsigned long long ga = (unsigned long long)gaddr;
  u32x4 g0;
  g0[0] = 1u;                                              // count=1, user mode
  g0[1] = lds_off;                                         // lds_addr (bytes)
  g0[2] = (unsigned)(ga & 0xFFFFFFFFu);                    // global_addr[31:0]
  g0[3] = (unsigned)((ga >> 32) & 0x01FFFFFFu) | (2u << 30); // global_addr[56:32], type=2
  i32x8 g1;
  g1[0] = (int)((1u << 16)                                 // data_size = 2 bytes
              | (1u << 20)                                 // pad_enable
              | (pad_interval_code << 22)
              | (pad_amount_code << 25));                  // workgroup_mask = 0 (no cluster)
  g1[1] = (int)((d0 & 0xFFFFu) << 16);                     // tensor_dim0[15:0] @ bit48
  g1[2] = (int)((d0 >> 16) | ((d1 & 0xFFFFu) << 16));      // tensor_dim0[31:16] | tensor_dim1[15:0]
  g1[3] = (int)((d1 >> 16) | ((d0 & 0xFFFFu) << 16));      // tensor_dim1[31:16] | tile_dim0
  g1[4] = (int)(d1 & 0xFFFFu);                             // tile_dim1 | tile_dim2=0
  g1[5] = (int)stride0;                                    // tensor_dim0_stride[31:0]
  g1[6] = 0;                                               // stride0[47:32] | stride1[15:0]
  g1[7] = 0;
  i32x4 z4 = {0, 0, 0, 0};
#if __has_include(<hip/amd_detail/amd_gfx1250_TDM.h>)
  i32x8 z8 = {0, 0, 0, 0, 0, 0, 0, 0};
  __builtin_amdgcn_tensor_load_to_lds(g0, g1, z4, z4, z8, 0);
#else
  __builtin_amdgcn_tensor_load_to_lds(g0, g1, z4, z4, 0);
#endif
}

// ============================================================
// Kernel 0: fp32 weights -> f16, pre-swizzled into B-operand tiles.
// ============================================================
__global__ void pack_w(const float* __restrict__ Wq, const float* __restrict__ Wk,
                       const float* __restrict__ Wv, _Float16* __restrict__ wp) {
  int gid = blockIdx.x * blockDim.x + threadIdx.x;
  if (gid >= 3 * D_ * D_) return;
  int wsel = gid >> 14;
  int r    = gid & 16383;
  int i    = r & 15;
  int lane = (r >> 4) & 31;
  int tile = r >> 9;
  int nt   = tile & 7;
  int kc   = tile >> 3;
  int h    = lane >> 4;
  int n    = nt * 16 + (lane & 15);
  int k    = kc * 32 + h * 8 + i + (i >> 3) * 8;
  const float* W = (wsel == 0) ? Wq : (wsel == 1) ? Wk : Wv;
  wp[gid] = (_Float16)W[k * D_ + n];
}

// ============================================================
// Kernel 1: Q/K/V projection (all WMMA). Q pre-scaled by (1/sqrt(D))*log2(e).
// Q,K row-major f16 [B*T][D]; V transposed f16 [B][D][T].
// ============================================================
__global__ __launch_bounds__(128) void qkv_proj(
    const float* __restrict__ x,
    const float* __restrict__ bq, const float* __restrict__ bk, const float* __restrict__ bv,
    const _Float16* __restrict__ wp,
    _Float16* __restrict__ Qh, _Float16* __restrict__ Kh, _Float16* __restrict__ Vt) {
  const int lane = threadIdx.x;
  const int tile = blockIdx.x * 4 + threadIdx.y;
  const int tok0 = tile * 16;
  const int m = lane & 15, h = lane >> 4;

  const float* xr = x + (size_t)(tok0 + m) * D_;
  v16h a[4];
#pragma unroll
  for (int kc = 0; kc < 4; kc++) {
#pragma unroll
    for (int g = 0; g < 2; g++) {
      const float* p = xr + kc * 32 + g * 16 + h * 8;
      float4 f0 = *(const float4*)p;
      float4 f1 = *(const float4*)(p + 4);
      a[kc][g * 8 + 0] = (_Float16)f0.x; a[kc][g * 8 + 1] = (_Float16)f0.y;
      a[kc][g * 8 + 2] = (_Float16)f0.z; a[kc][g * 8 + 3] = (_Float16)f0.w;
      a[kc][g * 8 + 4] = (_Float16)f1.x; a[kc][g * 8 + 5] = (_Float16)f1.y;
      a[kc][g * 8 + 6] = (_Float16)f1.z; a[kc][g * 8 + 7] = (_Float16)f1.w;
    }
  }
  const int b  = tok0 >> 12;
  const int t0 = tok0 & (T_ - 1);

#pragma unroll
  for (int osel = 0; osel < 3; osel++) {
    const float* bias = (osel == 0) ? bq : (osel == 1) ? bk : bv;
    const v16h* wt = (const v16h*)(wp + osel * (D_ * D_));
#pragma unroll
    for (int nt = 0; nt < 8; nt++) {
      float bvl = bias[nt * 16 + m];
      v8f acc;
#pragma unroll
      for (int j = 0; j < 8; j++) acc[j] = bvl;
#pragma unroll
      for (int kc = 0; kc < 4; kc++) {
        v16h w = wt[(kc * 8 + nt) * 32 + lane];
        acc = __builtin_amdgcn_wmma_f32_16x16x32_f16(false, a[kc], false, w,
                                                     (short)0, acc, false, false);
      }
      if (osel == 0) {
#pragma unroll
        for (int j = 0; j < 8; j++) acc[j] *= QSCALE;
      }
      if (osel < 2) {
        _Float16* dst = (osel == 0) ? Qh : Kh;
#pragma unroll
        for (int j = 0; j < 8; j++)
          dst[(size_t)(tok0 + j + 8 * h) * D_ + nt * 16 + m] = (_Float16)acc[j];
      } else {
        v8h pv;
#pragma unroll
        for (int j = 0; j < 8; j++) pv[j] = (_Float16)acc[j];
        *(v8h*)(Vt + (size_t)(b * D_ + nt * 16 + m) * T_ + t0 + 8 * h) = pv;
      }
    }
  }
}

// ============================================================
// Kernel 2: causal flash attention.
// Block = 4 waves = 64 queries; K/V tiles (32 keys) staged into LDS by the
// Tensor Data Mover, double-buffered, shared by all 4 waves.
// ============================================================
__global__ __launch_bounds__(128) void attn_fwd(
    const _Float16* __restrict__ Qh, const _Float16* __restrict__ Kh,
    const _Float16* __restrict__ Vt, float* __restrict__ out) {
  __shared__ _Float16 Kbuf[2][32 * KSTRIDE];    // 2 x 8704 B (incl. TDM pad)
  __shared__ _Float16 Vbuf[2][128 * VSTRIDE];   // 2 x 10240 B
  __shared__ _Float16 Pst[4][16 * PSTRIDE];     // per-wave P staging

  const int lane = threadIdx.x;
  const int wv   = threadIdx.y;
  const int qtile = blockIdx.x * 4 + wv;
  const int m = lane & 15, h = lane >> 4;
  const int tokq0 = qtile * 16;
  const int b  = tokq0 >> 12;
  const int q0 = tokq0 & (T_ - 1);

  const _Float16* Kb = Kh + (size_t)b * T_ * D_;
  const _Float16* Vb = Vt + (size_t)b * D_ * T_;

  // resident Q tile (A operands, already scaled into log2 domain)
  const _Float16* Qr = Qh + (size_t)(tokq0 + m) * D_;
  v16h aq[4];
#pragma unroll
  for (int kc = 0; kc < 4; kc++) aq[kc] = load_a16(Qr + kc * 32, h);

  v8f o[8];
#pragma unroll
  for (int nt = 0; nt < 8; nt++)
#pragma unroll
    for (int j = 0; j < 8; j++) o[nt][j] = 0.f;
  float mrow[8], lrow[8];
#pragma unroll
  for (int j = 0; j < 8; j++) { mrow[j] = -__builtin_inff(); lrow[j] = 0.f; }

  // shared trip count across the block (max diagonal of its 4 tiles)
  const int q0blk = (blockIdx.x * 64) & (T_ - 1);
  const int KBmax = (q0blk + 63) >> 5;

  // ---- stage key block kb into LDS buffer `buf` via TDM ----
  auto stage = [&](int kb, int buf) {
    const int k0 = kb * 32;
    tdm_load_2d((unsigned)(size_t)&Kbuf[buf][0], Kb + (size_t)k0 * D_,
                /*d0=*/128, /*d1=*/32, /*stride0=*/128, /*pi=*/5, /*pa=*/3);
    tdm_load_2d((unsigned)(size_t)&Vbuf[buf][0], Vb + k0,
                /*d0=*/32, /*d1=*/128, /*stride0=*/T_, /*pi=*/3, /*pa=*/3);
  };

  if (wv == 0) {
    stage(0, 0);
    __builtin_amdgcn_s_wait_tensorcnt(0);
  }
  __syncthreads();

  for (int kb = 0; kb <= KBmax; kb++) {
    const int k0 = kb * 32;
    if (wv == 0 && kb < KBmax) stage(kb + 1, (kb + 1) & 1);   // prefetch next tile

    const _Float16* Kt = &Kbuf[kb & 1][0];
    const _Float16* Vl = &Vbuf[kb & 1][0];

    // ---- S = Q K^T for 32 keys (two 16-key column tiles) ----
    v8f s0, s1;
#pragma unroll
    for (int j = 0; j < 8; j++) { s0[j] = 0.f; s1[j] = 0.f; }
#pragma unroll
    for (int kc = 0; kc < 4; kc++) {
      v16h k0op = load_b16(Kt + (size_t)m * KSTRIDE + kc * 32, h);
      s0 = __builtin_amdgcn_wmma_f32_16x16x32_f16(false, aq[kc], false, k0op,
                                                  (short)0, s0, false, false);
      v16h k1op = load_b16(Kt + (size_t)(16 + m) * KSTRIDE + kc * 32, h);
      s1 = __builtin_amdgcn_wmma_f32_16x16x32_f16(false, aq[kc], false, k1op,
                                                  (short)0, s1, false, false);
    }
    // ---- causal mask (any block reaching past this wave's diagonal) ----
    if (k0 + 31 > q0) {
#pragma unroll
      for (int j = 0; j < 8; j++) {
        int qi = q0 + j + 8 * h;
        if (k0 + m      > qi) s0[j] = -__builtin_inff();
        if (k0 + 16 + m > qi) s1[j] = -__builtin_inff();
      }
    }
    // ---- online softmax, log2 domain (rows in 16-lane halves: xor 1,2,4,8) ----
    float f[8];
#pragma unroll
    for (int j = 0; j < 8; j++) {
      float mx = fmaxf(s0[j], s1[j]);
#pragma unroll
      for (int d = 1; d < 16; d <<= 1) mx = fmaxf(mx, __shfl_xor(mx, d, 32));
      float mn = fmaxf(mrow[j], mx);
      f[j] = exp2f(mrow[j] - mn);
      mrow[j] = mn;
    }
    v8f p0, p1;
#pragma unroll
    for (int j = 0; j < 8; j++) {
      p0[j] = exp2f(s0[j] - mrow[j]);
      p1[j] = exp2f(s1[j] - mrow[j]);
      float rs = p0[j] + p1[j];
#pragma unroll
      for (int d = 1; d < 16; d <<= 1) rs += __shfl_xor(rs, d, 32);
      lrow[j] = lrow[j] * f[j] + rs;
    }
#pragma unroll
    for (int nt = 0; nt < 8; nt++)
#pragma unroll
      for (int j = 0; j < 8; j++) o[nt][j] *= f[j];

    // ---- P: C/D layout -> A layout via wave-private LDS tile ----
    _Float16* lp = Pst[wv];
#pragma unroll
    for (int j = 0; j < 8; j++) {
      lp[(j + 8 * h) * PSTRIDE + m]      = (_Float16)p0[j];
      lp[(j + 8 * h) * PSTRIDE + 16 + m] = (_Float16)p1[j];
    }
    asm volatile("s_wait_dscnt 0" ::: "memory");   // wave-local round-trip
    v16h ap = load_a16(lp + m * PSTRIDE, h);

    // ---- O += P V  (V tile in LDS: row = output dim, cols = 32 keys) ----
#pragma unroll
    for (int nt = 0; nt < 8; nt++) {
      v16h vop = load_b16(Vl + (size_t)(nt * 16 + m) * VSTRIDE, h);
      o[nt] = __builtin_amdgcn_wmma_f32_16x16x32_f16(false, ap, false, vop,
                                                     (short)0, o[nt], false, false);
    }

    if (wv == 0) __builtin_amdgcn_s_wait_tensorcnt(0);  // next tile resident
    __syncthreads();                                    // all waves done with this buf
  }

  // ---- epilogue: O / l, fp32 out [B*T][D] ----
  float inv[8];
#pragma unroll
  for (int j = 0; j < 8; j++) inv[j] = 1.0f / lrow[j];
#pragma unroll
  for (int nt = 0; nt < 8; nt++)
#pragma unroll
    for (int j = 0; j < 8; j++)
      out[(size_t)(tokq0 + j + 8 * h) * D_ + nt * 16 + m] = o[nt][j] * inv[j];
}

// ============================================================
extern "C" void kernel_launch(void* const* d_in, const int* in_sizes, int n_in,
                              void* d_out, int out_size, void* d_ws, size_t ws_size,
                              hipStream_t stream) {
  (void)in_sizes; (void)n_in; (void)out_size; (void)ws_size;
  const float* x  = (const float*)d_in[0];
  const float* Wq = (const float*)d_in[1];
  const float* bq = (const float*)d_in[2];
  const float* Wk = (const float*)d_in[3];
  const float* bk = (const float*)d_in[4];
  const float* Wv = (const float*)d_in[5];
  const float* bv = (const float*)d_in[6];

  char* ws = (char*)d_ws;
  _Float16* Qh = (_Float16*)(ws);                    // 4 MB  [B*T][D] f16 (log2-scaled)
  _Float16* Kh = (_Float16*)(ws + (4u  << 20));      // 4 MB  [B*T][D] f16
  _Float16* Vt = (_Float16*)(ws + (8u  << 20));      // 4 MB  [B][D][T] f16 (transposed)
  _Float16* Wp = (_Float16*)(ws + (12u << 20));      // 96 KB packed weights

  pack_w<<<dim3((3 * D_ * D_ + 255) / 256), dim3(256), 0, stream>>>(Wq, Wk, Wv, Wp);
  qkv_proj<<<dim3(256), dim3(32, 4), 0, stream>>>(x, bq, bk, bv, Wp, Qh, Kh, Vt);
  attn_fwd<<<dim3(256), dim3(32, 4), 0, stream>>>(Qh, Kh, Vt, (float*)d_out);
}
